// Encoder_56891136803037
// MI455X (gfx1250) — matrix operand order repaired and checked by reading, not compiled, not run on hardware
//
#include <hip/hip_runtime.h>
#include <hip/hip_bf16.h>
#include <math.h>

typedef __attribute__((ext_vector_type(16))) _Float16 v16h;
typedef __attribute__((ext_vector_type(8)))  _Float16 v8h;
typedef __attribute__((ext_vector_type(4)))  _Float16 v4h;
typedef __attribute__((ext_vector_type(8)))  float    v8f;
typedef __attribute__((ext_vector_type(4)))  int      v4i;

#define LDS_A_STRIDE 48    // 32 halves + 16 pad (rows 96B, 32B-aligned accesses)
#define LDS_B_STRIDE 144   // 128 halves + 16 pad (rows 288B, 32B-aligned accesses)
#define LDS_A_TILE (128 * LDS_A_STRIDE)
#define LDS_B_TILE (32 * LDS_B_STRIDE)

// ---------------------------------------------------------------------------
// CDNA5 async memory->LDS path (ASYNCcnt-tracked), with sync fallback.
// Builtin signature (from hipcc diagnostic): v4i ptr args, imm offset, imm cpol.
// ---------------------------------------------------------------------------
#if __has_builtin(__builtin_amdgcn_global_load_async_to_lds_b128)
#define ATTN_ASYNC_LDS 1
__device__ __forceinline__ void async_cp_b128(const _Float16* g, _Float16* l) {
  __builtin_amdgcn_global_load_async_to_lds_b128(
      (__attribute__((address_space(1))) v4i*)(void*)g,
      (__attribute__((address_space(3))) v4i*)(void*)l,
      /*offset=*/0, /*cpol=*/0);
}
#endif

template <int N>
__device__ __forceinline__ void wait_async_le() {
#ifdef ATTN_ASYNC_LDS
#if __has_builtin(__builtin_amdgcn_s_wait_asynccnt)
  __builtin_amdgcn_s_wait_asynccnt(N);
#else
  asm volatile("s_wait_asynccnt %0" ::"n"(N) : "memory");
#endif
#endif
}

// ---------------------------------------------------------------------------
// Fragment loads from LDS, matching CDNA5 WMMA VGPR layouts (CDNA5 ISA §7.12.2)
// A 16x32 f16: lanes 0-15 hold row m, K chunks [0..7],[16..23];
//              lanes 16-31 hold row m, K chunks [8..15],[24..31].
// B 32x16 f16: lane L holds column N=L&15, same K chunks as A.
// ---------------------------------------------------------------------------
__device__ __forceinline__ v16h frag_a_lds(const _Float16* As, int lane) {
  const int m   = lane & 15;
  const int sel = lane >> 4;
  const _Float16* p = As + m * LDS_A_STRIDE + sel * 8;
  v8h lo = *(const v8h*)(p);        // K = sel*8 .. sel*8+7
  v8h hi = *(const v8h*)(p + 16);   // K = 16+sel*8 .. 16+sel*8+7
  v16h f;
#pragma unroll
  for (int i = 0; i < 8; ++i) { f[i] = lo[i]; f[i + 8] = hi[i]; }
  return f;
}

__device__ __forceinline__ v16h frag_b_lds(const _Float16* Bs, int lane) {
  const int n   = lane & 15;
  const int sel = lane >> 4;
  v16h f;
#pragma unroll
  for (int i = 0; i < 8; ++i) {
    f[i]     = Bs[(sel * 8 + i) * LDS_B_STRIDE + n];
    f[i + 8] = Bs[(16 + sel * 8 + i) * LDS_B_STRIDE + n];
  }
  return f;
}

// ---------------------------------------------------------------------------
// Stage one 128x32 A tile + 32x128 B tile into an LDS buffer.
// Async path: 8 x global_load_async_to_lds_b128 per thread (ASYNCcnt += 8/wave).
// ---------------------------------------------------------------------------
__device__ __forceinline__ void stage_tiles(
    const _Float16* __restrict__ Ag, int lda,
    const _Float16* __restrict__ Bg, int ldb,
    int k0, int blockRow, int blockCol,
    _Float16* As, _Float16* Bs, int t)
{
  const _Float16* ga = Ag + (size_t)(blockRow + t) * lda + k0;
  const int br = t >> 2, bc = (t & 3) * 32;
  const _Float16* gb = Bg + (size_t)(k0 + br) * ldb + blockCol + bc;
  _Float16* la = As + t * LDS_A_STRIDE;
  _Float16* lb = Bs + br * LDS_B_STRIDE + bc;
#ifdef ATTN_ASYNC_LDS
  async_cp_b128(ga,      la);
  async_cp_b128(ga + 8,  la + 8);
  async_cp_b128(ga + 16, la + 16);
  async_cp_b128(ga + 24, la + 24);
  async_cp_b128(gb,      lb);
  async_cp_b128(gb + 8,  lb + 8);
  async_cp_b128(gb + 16, lb + 16);
  async_cp_b128(gb + 24, lb + 24);
#else
  v16h a0 = *(const v16h*)(ga);
  v16h a1 = *(const v16h*)(ga + 16);
  v16h b0 = *(const v16h*)(gb);
  v16h b1 = *(const v16h*)(gb + 16);
  *(v16h*)(la)      = a0;
  *(v16h*)(la + 16) = a1;
  *(v16h*)(lb)      = b0;
  *(v16h*)(lb + 16) = b1;
#endif
}

// ---------------------------------------------------------------------------
// 128x128 block tile GEMM core: C(MxN) += A(MxK) * B(KxN), f16 in, f32 acc.
// 4 waves (128 threads), each wave computes a 64x64 sub-tile (16 WMMA tiles).
// Double-buffered LDS; async loads for buffer k+1 overlap WMMAs on buffer k.
// ---------------------------------------------------------------------------
__device__ __forceinline__ void gemm_core_128x128(
    const _Float16* __restrict__ Ag, int lda,
    const _Float16* __restrict__ Bg, int ldb,
    int K, int blockRow, int blockCol,
    _Float16* AsBase, _Float16* BsBase, v8f acc[4][4])
{
  const int t    = threadIdx.x;       // 0..127
  const int lane = t & 31;
  const int w    = t >> 5;
  const int wm   = (w & 1) * 64;
  const int wn   = (w >> 1) * 64;

  const v8f zero = {0.f, 0.f, 0.f, 0.f, 0.f, 0.f, 0.f, 0.f};
#pragma unroll
  for (int i = 0; i < 4; ++i)
#pragma unroll
    for (int j = 0; j < 4; ++j) acc[i][j] = zero;

  const int nk = K >> 5;
  stage_tiles(Ag, lda, Bg, ldb, 0, blockRow, blockCol, AsBase, BsBase, t);

  for (int ik = 0; ik < nk; ++ik) {
    _Float16* As = AsBase + (ik & 1) * LDS_A_TILE;
    _Float16* Bs = BsBase + (ik & 1) * LDS_B_TILE;

    if (ik + 1 < nk) {
      // Buffer (ik+1)&1 was last read in iteration ik-1, which all waves
      // exited through a barrier -> safe to overwrite now.
      stage_tiles(Ag, lda, Bg, ldb, (ik + 1) << 5, blockRow, blockCol,
                  AsBase + ((ik + 1) & 1) * LDS_A_TILE,
                  BsBase + ((ik + 1) & 1) * LDS_B_TILE, t);
      wait_async_le<8>();   // the 8 ops for the CURRENT buffer have retired
    } else {
      wait_async_le<0>();
    }
    __syncthreads();        // every wave's staging of current buffer visible

    v16h af[4], bf[4];
#pragma unroll
    for (int i = 0; i < 4; ++i)
      af[i] = frag_a_lds(As + (wm + 16 * i) * LDS_A_STRIDE, lane);
#pragma unroll
    for (int j = 0; j < 4; ++j)
      bf[j] = frag_b_lds(Bs + wn + 16 * j, lane);

#pragma unroll
    for (int i = 0; i < 4; ++i)
#pragma unroll
      for (int j = 0; j < 4; ++j)
        acc[i][j] = __builtin_amdgcn_wmma_f32_16x16x32_f16(
            false, af[i], false, bf[j], (short)0, acc[i][j], false, false);
    __syncthreads();        // all reads of current buffer done
  }
}

// D-layout (16x16 f32): VGPR j -> row (j + 8*(lane>>4)), col = lane&15.
#define EPILOGUE_INDICES \
  const int t = threadIdx.x, lane = t & 31, w = t >> 5;   \
  const int wm = (w & 1) * 64, wn = (w >> 1) * 64;        \
  const int nl = lane & 15, mb = (lane >> 4) * 8;

#define DECLARE_GEMM_LDS \
  __shared__ __align__(32) _Float16 As[2 * LDS_A_TILE]; \
  __shared__ __align__(32) _Float16 Bs[2 * LDS_B_TILE];

// ---------------------------------------------------------------------------
// Kernel 1: f32 -> f16 bulk convert
// ---------------------------------------------------------------------------
__global__ void cvt_f32_f16_kernel(const float* __restrict__ in,
                                   _Float16* __restrict__ out, size_t n) {
  size_t i = ((size_t)blockIdx.x * blockDim.x + threadIdx.x) * 4;
  if (i >= n) return;
  float4 f = *(const float4*)(in + i);
  v4h h; h[0] = (_Float16)f.x; h[1] = (_Float16)f.y;
         h[2] = (_Float16)f.z; h[3] = (_Float16)f.w;
  *(v4h*)(out + i) = h;
}

// ---------------------------------------------------------------------------
// Kernel 2: W[n][k] f32 -> Out[k][n] f16 (so B-operand rows are contiguous)
// ---------------------------------------------------------------------------
__global__ void transpose_w_kernel(const float* __restrict__ W,
                                   _Float16* __restrict__ out, int ldo) {
  int n = blockIdx.x * 32 + (threadIdx.x & 31);
  int k = blockIdx.y * 8 + (threadIdx.x >> 5);
  out[(size_t)k * ldo + n] = (_Float16)W[(size_t)n * 1024 + k];
}

// ---------------------------------------------------------------------------
// Kernel 3: fused QKV projection.  QKV = Xh[b] @ Wcat (1024 x 3072).
// Epilogue: Q scaled by 1/sqrt(1024); K stored transposed [d][s]; V row-major.
// ---------------------------------------------------------------------------
__global__ void __launch_bounds__(128)
qkv_gemm_kernel(const _Float16* __restrict__ Xh, const _Float16* __restrict__ Wcat,
                _Float16* __restrict__ Qh, _Float16* __restrict__ Kt,
                _Float16* __restrict__ Vh) {
  DECLARE_GEMM_LDS
  const int b = blockIdx.z;
  const int blockRow = blockIdx.x * 128, blockCol = blockIdx.y * 128;
  v8f acc[4][4];
  gemm_core_128x128(Xh + (size_t)b * 2048 * 1024, 1024, Wcat, 3072, 1024,
                    blockRow, blockCol, As, Bs, acc);
  EPILOGUE_INDICES
  const size_t qb = (size_t)b * 2048 * 1024;
#pragma unroll
  for (int i = 0; i < 4; ++i)
#pragma unroll
    for (int j = 0; j < 4; ++j)
#pragma unroll
      for (int r = 0; r < 8; ++r) {
        int gm = blockRow + wm + i * 16 + mb + r;
        int gn = blockCol + wn + j * 16 + nl;
        float v = acc[i][j][r];
        if (gn < 1024)                 // Q, pre-scaled by 1/sqrt(dk)
          Qh[qb + (size_t)gm * 1024 + gn] = (_Float16)(v * 0.03125f);
        else if (gn < 2048)            // K, stored transposed [d][s]
          Kt[qb + (size_t)(gn - 1024) * 2048 + gm] = (_Float16)v;
        else                           // V, row-major
          Vh[qb + (size_t)gm * 1024 + (gn - 2048)] = (_Float16)v;
      }
}

// ---------------------------------------------------------------------------
// Kernel 4: scores S = Q @ Kt   (M=N=2048, K=1024), f16 store
// ---------------------------------------------------------------------------
__global__ void __launch_bounds__(128)
scores_gemm_kernel(const _Float16* __restrict__ Qh, const _Float16* __restrict__ Kt,
                   _Float16* __restrict__ S) {
  DECLARE_GEMM_LDS
  const int b = blockIdx.z;
  const int blockRow = blockIdx.x * 128, blockCol = blockIdx.y * 128;
  v8f acc[4][4];
  gemm_core_128x128(Qh + (size_t)b * 2048 * 1024, 1024,
                    Kt + (size_t)b * 1024 * 2048, 2048, 1024,
                    blockRow, blockCol, As, Bs, acc);
  EPILOGUE_INDICES
  _Float16* Sb = S + (size_t)b * 2048 * 2048;
#pragma unroll
  for (int i = 0; i < 4; ++i)
#pragma unroll
    for (int j = 0; j < 4; ++j)
#pragma unroll
      for (int r = 0; r < 8; ++r) {
        int gm = blockRow + wm + i * 16 + mb + r;
        int gn = blockCol + wn + j * 16 + nl;
        Sb[(size_t)gm * 2048 + gn] = (_Float16)acc[i][j][r];
      }
}

// ---------------------------------------------------------------------------
// Kernel 5: per-row max and 1/sum(exp) over 2048 scores. One wave32 per row.
// ---------------------------------------------------------------------------
__global__ void __launch_bounds__(256)
rowstats_kernel(const _Float16* __restrict__ S, float* __restrict__ mrow,
                float* __restrict__ linv) {
  const int row  = blockIdx.x * 8 + (threadIdx.x >> 5);   // 16384 rows total
  const int lane = threadIdx.x & 31;
  const _Float16* p = S + (size_t)row * 2048;
  float v[64];
  float mx = -3.0e38f;
#pragma unroll
  for (int i = 0; i < 64; ++i) {
    v[i] = (float)p[lane + i * 32];
    mx = fmaxf(mx, v[i]);
  }
#pragma unroll
  for (int off = 16; off > 0; off >>= 1)
    mx = fmaxf(mx, __shfl_xor(mx, off, 32));
  float s = 0.f;
#pragma unroll
  for (int i = 0; i < 64; ++i) s += __expf(v[i] - mx);
#pragma unroll
  for (int off = 16; off > 0; off >>= 1)
    s += __shfl_xor(s, off, 32);
  if (lane == 0) { mrow[row] = mx; linv[row] = 1.0f / s; }
}

// ---------------------------------------------------------------------------
// Kernel 6: in-place P = exp(S - rowmax), f16. 8 halves / thread.
// ---------------------------------------------------------------------------
__global__ void __launch_bounds__(256)
expnorm_kernel(_Float16* __restrict__ S, const float* __restrict__ mrow) {
  size_t idx = ((size_t)blockIdx.x * 256 + threadIdx.x) * 8;
  const float m = mrow[idx >> 11];         // row = idx / 2048
  v8h d = *(v8h*)(S + idx);
#pragma unroll
  for (int i = 0; i < 8; ++i) d[i] = (_Float16)__expf((float)d[i] - m);
  *(v8h*)(S + idx) = d;
}

// ---------------------------------------------------------------------------
// Kernel 7: ctx = (P @ V) * (1/l) rowwise  (M=2048, N=1024, K=2048), f16 store
// ---------------------------------------------------------------------------
__global__ void __launch_bounds__(128)
ctx_gemm_kernel(const _Float16* __restrict__ P, const _Float16* __restrict__ Vh,
                const float* __restrict__ linv, _Float16* __restrict__ Ctx) {
  DECLARE_GEMM_LDS
  const int b = blockIdx.z;
  const int blockRow = blockIdx.x * 128, blockCol = blockIdx.y * 128;
  v8f acc[4][4];
  gemm_core_128x128(P + (size_t)b * 2048 * 2048, 2048,
                    Vh + (size_t)b * 2048 * 1024, 1024, 2048,
                    blockRow, blockCol, As, Bs, acc);
  EPILOGUE_INDICES
  _Float16* Cb = Ctx + (size_t)b * 2048 * 1024;
  const float* lb = linv + (size_t)b * 2048;
#pragma unroll
  for (int i = 0; i < 4; ++i)
#pragma unroll
    for (int j = 0; j < 4; ++j)
#pragma unroll
      for (int r = 0; r < 8; ++r) {
        int gm = blockRow + wm + i * 16 + mb + r;
        int gn = blockCol + wn + j * 16 + nl;
        Cb[(size_t)gm * 1024 + gn] = (_Float16)(acc[i][j][r] * lb[gm]);
      }
}

// ---------------------------------------------------------------------------
// Kernel 8: out = ctx @ Wo^T  (M=2048, N=1024, K=1024), f32 store
// ---------------------------------------------------------------------------
__global__ void __launch_bounds__(128)
out_gemm_kernel(const _Float16* __restrict__ Ctx, const _Float16* __restrict__ Wot,
                float* __restrict__ Out) {
  DECLARE_GEMM_LDS
  const int b = blockIdx.z;
  const int blockRow = blockIdx.x * 128, blockCol = blockIdx.y * 128;
  v8f acc[4][4];
  gemm_core_128x128(Ctx + (size_t)b * 2048 * 1024, 1024, Wot, 1024, 1024,
                    blockRow, blockCol, As, Bs, acc);
  EPILOGUE_INDICES
  float* Ob = Out + (size_t)b * 2048 * 1024;
#pragma unroll
  for (int i = 0; i < 4; ++i)
#pragma unroll
    for (int j = 0; j < 4; ++j)
#pragma unroll
      for (int r = 0; r < 8; ++r) {
        int gm = blockRow + wm + i * 16 + mb + r;
        int gn = blockCol + wn + j * 16 + nl;
        Ob[(size_t)gm * 1024 + gn] = acc[i][j][r];
      }
}

// ---------------------------------------------------------------------------
// Host orchestration
// ---------------------------------------------------------------------------
extern "C" void kernel_launch(void* const* d_in, const int* in_sizes, int n_in,
                              void* d_out, int out_size, void* d_ws, size_t ws_size,
                              hipStream_t stream) {
  const float* x  = (const float*)d_in[0];
  const float* wq = (const float*)d_in[1];
  const float* wk = (const float*)d_in[2];
  const float* wv = (const float*)d_in[3];
  const float* wo = (const float*)d_in[4];

  char* ws = (char*)d_ws;
  const size_t XN = (size_t)8 * 2048 * 1024;        // 16,777,216 elems
  _Float16* Xh   = (_Float16*)(ws);                                 // 33.55 MB
  _Float16* Wcat = (_Float16*)(ws + 33554432);                      //  6.29 MB
  _Float16* Wot  = (_Float16*)(ws + 39845888);                      //  2.10 MB
  _Float16* Qh   = (_Float16*)(ws + 41943040);                      // 33.55 MB
  _Float16* Kt   = (_Float16*)(ws + 75497472);                      // 33.55 MB
  _Float16* Vh   = (_Float16*)(ws + 109051904);                     // 33.55 MB
  _Float16* S    = (_Float16*)(ws + 142606336);                     // 67.11 MB
  float*    mrow = (float*)   (ws + 209715200);                     // 64 KB
  float*    linv = (float*)   (ws + 209780736);                     // 64 KB
  _Float16* Ctx  = Xh;   // x (f16) dead after QKV projection; reuse buffer

  // 1) convert x to f16
  cvt_f32_f16_kernel<<<(unsigned)(XN / (4 * 256)), 256, 0, stream>>>(x, Xh, XN);
  // 2) transpose weights into f16 B-operand layout
  transpose_w_kernel<<<dim3(32, 128), 256, 0, stream>>>(wq, Wcat + 0,    3072);
  transpose_w_kernel<<<dim3(32, 128), 256, 0, stream>>>(wk, Wcat + 1024, 3072);
  transpose_w_kernel<<<dim3(32, 128), 256, 0, stream>>>(wv, Wcat + 2048, 3072);
  transpose_w_kernel<<<dim3(32, 128), 256, 0, stream>>>(wo, Wot,         1024);
  // 3) fused QKV projection (per batch 2048x3072)
  qkv_gemm_kernel<<<dim3(16, 24, 8), 128, 0, stream>>>(Xh, Wcat, Qh, Kt, Vh);
  // 4) scores S = Q @ Kt
  scores_gemm_kernel<<<dim3(16, 16, 8), 128, 0, stream>>>(Qh, Kt, S);
  // 5) row max + 1/sum(exp)
  rowstats_kernel<<<2048, 256, 0, stream>>>(S, mrow, linv);
  // 6) P = exp(S - m) in place
  expnorm_kernel<<<(unsigned)((size_t)8 * 2048 * 2048 / (8 * 256)), 256, 0, stream>>>(S, mrow);
  // 7) ctx = (P @ V) / l
  ctx_gemm_kernel<<<dim3(16, 8, 8), 128, 0, stream>>>(S, Vh, linv, Ctx);
  // 8) out = ctx @ Wo^T (f32)
  out_gemm_kernel<<<dim3(16, 8, 8), 128, 0, stream>>>(Ctx, Wot, (float*)d_out);
}
